// KANLayer_61538291417474
// MI455X (gfx1250) — matrix operand order repaired and checked by reading, not compiled, not run on hardware
//
#include <hip/hip_runtime.h>

// Problem dims (match reference)
#define BDIM 4096   // batch (GEMM M)
#define DDIM 1024   // input dim
#define KBAS 5      // num polynomial basis
#define UDIM 1024   // units (GEMM N)
#define KTOT (DDIM * KBAS)  // 5120 (GEMM K)

// Tiling
#define BM 128
#define BN 64
#define DI 32                 // x-columns staged per outer iter
#define BK (DI * KBAS)        // 160 = 5 WMMA K-steps of 32
#define AS 168                // LDS A row stride in halves (160 + 8 pad, keeps 16B align)
#define BS 168                // LDS B col stride in halves
#define NTHREADS 256

typedef __attribute__((ext_vector_type(16))) _Float16 v16h;
typedef __attribute__((ext_vector_type(8)))  _Float16 v8h;
typedef __attribute__((ext_vector_type(8)))  float    v8f;

union Frag16 {
    v16h v;
    v8h  h[2];
};

__global__ __launch_bounds__(NTHREADS) void kan_wmma_gemm(
    const float* __restrict__ x,      // (BDIM, DDIM)
    const float* __restrict__ w,      // (DDIM, KBAS, UDIM) == row-major (KTOT, UDIM)
    const float* __restrict__ bias,   // (UDIM)
    float* __restrict__ out)          // (BDIM, UDIM)
{
    __shared__ _Float16 ldsA[BM * AS];  // 43008 B
    __shared__ _Float16 ldsB[BN * BS];  // 21504 B  (total 64512 B < 64KB)

    const int tid  = threadIdx.x;
    const int lane = tid & 31;
    const int wave = tid >> 5;
    const int wm   = wave >> 1;   // 0..3  (rows of 32)
    const int wn   = wave & 1;    // 0..1  (cols of 32)
    const int m0   = blockIdx.x * BM;
    const int n0   = blockIdx.y * BN;

    const int l16     = lane & 15;
    const int halfsel = lane >> 4;   // 0 or 1

    v8f acc[2][2] = {};

    for (int tt = 0; tt < DDIM / DI; ++tt) {
        const int i0  = tt * DI;    // x column base
        const int kk0 = tt * BK;    // flattened-K base

        __syncthreads();  // protect LDS WAR vs previous compute phase

        // ---- Stage A: build x^k basis tile (BM x BK) in f16, row-major ----
        #pragma unroll
        for (int s = 0; s < (BM * DI) / NTHREADS; ++s) {
            int p  = tid + s * NTHREADS;
            int r  = p >> 5;        // row within tile (lane group covers a row)
            int ii = p & 31;        // x column within chunk
            float xv = x[(m0 + r) * DDIM + (i0 + ii)];
            float x2 = xv * xv;
            _Float16* dst = &ldsA[r * AS + ii * KBAS];
            dst[0] = (_Float16)1.0f;
            dst[1] = (_Float16)xv;
            dst[2] = (_Float16)x2;
            dst[3] = (_Float16)(x2 * xv);
            dst[4] = (_Float16)(x2 * x2);
        }

        // ---- Stage B: weights tile (BK x BN) -> LDS column-major f16 ----
        #pragma unroll 8
        for (int s = 0; s < (BK * BN) / NTHREADS; ++s) {
            int p   = tid + s * NTHREADS;
            int n   = p & (BN - 1);   // coalesced over n
            int kkl = p >> 6;
            float wv = w[(kk0 + kkl) * UDIM + (n0 + n)];
            ldsB[n * BS + kkl] = (_Float16)wv;
        }

        // Prefetch next chunk (gfx1250 global_prefetch_b8)
        if (tt + 1 < DDIM / DI) {
            __builtin_prefetch(&x[(m0 + (tid >> 5)) * DDIM + i0 + DI + (tid & 31)], 0, 0);
            __builtin_prefetch(&w[(kk0 + BK + (tid >> 6)) * UDIM + n0 + (tid & 63)], 0, 0);
        }

        __syncthreads();

        // ---- Compute: 5 WMMA K-steps over the staged BK=160 ----
        #pragma unroll
        for (int ks = 0; ks < KBAS; ++ks) {
            Frag16 af[2], bf[2];
            // A fragment: lane L holds row L%16; K = (L/16)*8 + {0..7, 16..23}
            #pragma unroll
            for (int mt = 0; mt < 2; ++mt) {
                int r  = wm * 32 + mt * 16 + l16;
                int h0 = r * AS + ks * 32 + halfsel * 8;
                af[mt].h[0] = *(const v8h*)&ldsA[h0];
                af[mt].h[1] = *(const v8h*)&ldsA[h0 + 16];
            }
            // B fragment: lane L holds col L%16; K = (L/16)*8 + {0..7, 16..23}
            #pragma unroll
            for (int nt = 0; nt < 2; ++nt) {
                int n  = wn * 32 + nt * 16 + l16;
                int h0 = n * BS + ks * 32 + halfsel * 8;
                bf[nt].h[0] = *(const v8h*)&ldsB[h0];
                bf[nt].h[1] = *(const v8h*)&ldsB[h0 + 16];
            }
            #pragma unroll
            for (int mt = 0; mt < 2; ++mt)
                #pragma unroll
                for (int nt = 0; nt < 2; ++nt)
                    acc[mt][nt] = __builtin_amdgcn_wmma_f32_16x16x32_f16(
                        false, af[mt].v, false, bf[nt].v,
                        (short)0, acc[mt][nt], false, false);
        }
    }

    // ---- Epilogue: bias + exact GELU, C layout: M = v + 8*(lane/16), N = lane%16 ----
    #pragma unroll
    for (int nt = 0; nt < 2; ++nt) {
        int gn = n0 + wn * 32 + nt * 16 + l16;
        float bv = bias[gn];
        #pragma unroll
        for (int mt = 0; mt < 2; ++mt) {
            #pragma unroll
            for (int v = 0; v < 8; ++v) {
                int gm = m0 + wm * 32 + mt * 16 + v + 8 * halfsel;
                float val = acc[mt][nt][v] + bv;
                float g = 0.5f * val * (1.0f + erff(val * 0.70710678118654752f));
                out[gm * UDIM + gn] = g;
            }
        }
    }
}

extern "C" void kernel_launch(void* const* d_in, const int* in_sizes, int n_in,
                              void* d_out, int out_size, void* d_ws, size_t ws_size,
                              hipStream_t stream) {
    (void)in_sizes; (void)n_in; (void)d_ws; (void)ws_size; (void)out_size;
    const float* x    = (const float*)d_in[0];  // (4096, 1024)
    const float* w    = (const float*)d_in[1];  // (1024, 5, 1024)
    const float* bias = (const float*)d_in[2];  // (1024,)
    float* out        = (float*)d_out;          // (4096, 1024)

    dim3 grid(BDIM / BM, UDIM / BN);  // (32, 16)
    dim3 block(NTHREADS);
    kan_wmma_gemm<<<grid, block, 0, stream>>>(x, w, bias, out);
}
